// MambaLayer_103079215437
// MI455X (gfx1250) — matrix operand order repaired and checked by reading, not compiled, not run on hardware
//
#include <hip/hip_runtime.h>
#include <hip/hip_bf16.h>
#include <stdint.h>

// ---------------- problem constants ----------------
#define D_MODEL   1024
#define D_STATE   32
#define D_CONV    4
#define D_INNER   2048          // EXPAND * D_MODEL
#define D_OUT     128
#define BATCH     8
#define SEQ       4096
#define NPOS      (BATCH * SEQ) // 32768
#define N_SSM     65            // 2*D_STATE + 1
#define N_SSM_PAD 128           // padded to block-tile width

// GEMM tiling
#define M_BLK     128
#define N_BLK     128
#define K_STEP    32
#define LDS_LD    40            // 32 data + 8 pad elems; 80B row stride (16B aligned, conflict-free)
#define TILE_ELMS (M_BLK * LDS_LD)

typedef __attribute__((ext_vector_type(16))) __bf16 v16bf;
typedef __attribute__((ext_vector_type(8)))  float  v8f;

// float -> bf16 (round to nearest even), raw u16 storage
__device__ __forceinline__ uint16_t f2bf(float f) {
    unsigned u = __float_as_uint(f);
    u += 0x7FFFu + ((u >> 16) & 1u);
    return (uint16_t)(u >> 16);
}

// ---- CDNA5 async global->LDS copy (ASYNCcnt tracked, no VGPR staging) ----
// GLOBAL_LOAD_ASYNC_TO_LDS_B128: VDST = per-lane LDS byte address (low 32 bits of
// the flat pointer, per ISA LDS aperture mapping), VADDR = 64-bit global address.
__device__ __forceinline__ void async_b128(void* lds_ptr, const void* gptr) {
    const uint32_t loff = (uint32_t)(uintptr_t)lds_ptr;
    const unsigned long long gaddr = (unsigned long long)(uintptr_t)gptr;
    asm volatile("global_load_async_to_lds_b128 %0, %1, off"
                 :: "v"(loff), "v"(gaddr)
                 : "memory");
}
__device__ __forceinline__ void wait_async0() {
    asm volatile("s_wait_asynccnt 0x0" ::: "memory");
}

// ---- WMMA fragment loaders from LDS (ISA 7.12.2, wave32) ----
// A 16x32 bf16: lane l -> row m=l&15; half h=l>>4 holds K = h*8+0..7 and 16+h*8+0..7
__device__ __forceinline__ v16bf lds_frag_a(const uint16_t* __restrict__ s, int row0, int lane) {
    const int m = lane & 15, h = lane >> 4;
    const uint16_t* p = s + (row0 + m) * LDS_LD + h * 8;
    union { v16bf v; uint4 u[2]; } r;
    r.u[0] = *(const uint4*)(p);
    r.u[1] = *(const uint4*)(p + 16);
    return r.v;
}
// B 32x16 bf16 (one W-row == one B-column per lane): lane l -> col n=l&15,
// half h=l>>4 holds K = h*8+0..7 and 16+h*8+0..7
__device__ __forceinline__ v16bf lds_frag_b(const uint16_t* __restrict__ s, int row0, int lane) {
    const int n = lane & 15, h = lane >> 4;
    const uint16_t* p = s + (row0 + n) * LDS_LD + h * 8;
    union { v16bf v; uint4 u[2]; } r;
    r.u[0] = *(const uint4*)(p);
    r.u[1] = *(const uint4*)(p + 16);
    return r.v;
}
// D/C 16x16 f32: lane l -> col n=l&15, VGPR r -> row r + 8*(l>>4)

// ---------------- shared GEMM core ----------------
// Block = 256 thr = 8 waves (4 in M x 2 in N); block tile 128x128; wave tile 32x64.
// Double-buffered LDS, filled by async DMA (global_load_async_to_lds_b128): the
// async copies for tile k+1 are issued, then the wave immediately runs the
// 12 ds_load_b128 + 8 v_wmma of tile k. s_wait_asynccnt 0 + barrier per K-step.
__device__ __forceinline__ void gemm_core(
    const uint16_t* __restrict__ A,   // M x K bf16 (row-major)
    const uint16_t* __restrict__ W,   // N x K bf16 (row-major)
    int K, int m0, int n0,
    uint16_t* __restrict__ smem,      // 4 * TILE_ELMS u16
    v8f acc[2][4], int lane, int wm, int wn)
{
    const int tid = threadIdx.x;
    uint16_t* As[2] = { smem,                 smem + TILE_ELMS };
    uint16_t* Bs[2] = { smem + 2 * TILE_ELMS, smem + 3 * TILE_ELMS };

    auto async_tile = [&](int k0, int buf) {
        #pragma unroll
        for (int i = 0; i < 2; ++i) {
            const int u   = tid + i * 256;      // 0..511 : 4 x uint4 per 32-elem row
            const int row = u >> 2;
            const int c8  = (u & 3) * 8;
            async_b128(As[buf] + row * LDS_LD + c8, A + (size_t)(m0 + row) * K + k0 + c8);
            async_b128(Bs[buf] + row * LDS_LD + c8, W + (size_t)(n0 + row) * K + k0 + c8);
        }
    };
    auto mma = [&](int buf) {
        const uint16_t* sa = As[buf];
        const uint16_t* sb = Bs[buf];
        v16bf a0 = lds_frag_a(sa, wm * 32,      lane);
        v16bf a1 = lds_frag_a(sa, wm * 32 + 16, lane);
        v16bf b0 = lds_frag_b(sb, wn * 64 +  0, lane);
        v16bf b1 = lds_frag_b(sb, wn * 64 + 16, lane);
        v16bf b2 = lds_frag_b(sb, wn * 64 + 32, lane);
        v16bf b3 = lds_frag_b(sb, wn * 64 + 48, lane);
        acc[0][0] = __builtin_amdgcn_wmma_f32_16x16x32_bf16(false, a0, false, b0, (short)0, acc[0][0], false, false);
        acc[1][0] = __builtin_amdgcn_wmma_f32_16x16x32_bf16(false, a1, false, b0, (short)0, acc[1][0], false, false);
        acc[0][1] = __builtin_amdgcn_wmma_f32_16x16x32_bf16(false, a0, false, b1, (short)0, acc[0][1], false, false);
        acc[1][1] = __builtin_amdgcn_wmma_f32_16x16x32_bf16(false, a1, false, b1, (short)0, acc[1][1], false, false);
        acc[0][2] = __builtin_amdgcn_wmma_f32_16x16x32_bf16(false, a0, false, b2, (short)0, acc[0][2], false, false);
        acc[1][2] = __builtin_amdgcn_wmma_f32_16x16x32_bf16(false, a1, false, b2, (short)0, acc[1][2], false, false);
        acc[0][3] = __builtin_amdgcn_wmma_f32_16x16x32_bf16(false, a0, false, b3, (short)0, acc[0][3], false, false);
        acc[1][3] = __builtin_amdgcn_wmma_f32_16x16x32_bf16(false, a1, false, b3, (short)0, acc[1][3], false, false);
    };

    async_tile(0, 0);
    wait_async0();
    __syncthreads();
    int buf = 0;
    for (int k = K_STEP; k < K; k += K_STEP) {
        async_tile(k, buf ^ 1);   // DMA next tile straight into LDS (no VGPRs)
        mma(buf);                 // overlaps with the async copy
        wait_async0();            // own transfers landed in LDS
        __syncthreads();
        buf ^= 1;
    }
    mma(buf);
}

// ---------------- conversion kernels ----------------
__global__ void k_f2bf(const float* __restrict__ src, uint16_t* __restrict__ dst, int n) {
    int i = blockIdx.x * 256 + threadIdx.x;
    if (i < n) dst[i] = f2bf(src[i]);
}

__global__ void k_pad_wx(const float* __restrict__ Wx, const float* __restrict__ bx,
                         uint16_t* __restrict__ Wp, float* __restrict__ bp) {
    int i = blockIdx.x * 256 + threadIdx.x;
    if (i < N_SSM_PAD * D_INNER) {
        int r = i / D_INNER;
        Wp[i] = (r < N_SSM) ? f2bf(Wx[i]) : (uint16_t)0;
    }
    if (i < N_SSM_PAD) bp[i] = (i < N_SSM) ? bx[i] : 0.f;
}

__global__ void k_rowsum(const float* __restrict__ Wout, float* __restrict__ rowsum) {
    int o = threadIdx.x;
    if (o < D_OUT) {
        float s = 0.f;
        for (int i = 0; i < D_INNER; ++i) s += Wout[(size_t)o * D_INNER + i];
        rowsum[o] = s;
    }
}

// ---------------- GEMM1: xr = x @ W_in^T + b_in ; split -> xs(f32), res(bf16) ----------------
__global__ __launch_bounds__(256) void k_gemm1(
    const uint16_t* __restrict__ A,   // NPOS x D_MODEL bf16
    const uint16_t* __restrict__ W,   // 2*D_INNER x D_MODEL bf16
    const float*    __restrict__ bias,
    float*          __restrict__ xs,  // NPOS x D_INNER f32
    uint16_t*       __restrict__ res) // NPOS x D_INNER bf16
{
    __shared__ uint16_t smem[4 * TILE_ELMS];
    const int lane = threadIdx.x & 31, wave = threadIdx.x >> 5;
    const int wm = wave & 3, wn = wave >> 2;
    const int m0 = blockIdx.y * M_BLK, n0 = blockIdx.x * N_BLK;
    v8f acc[2][4] = {};
    gemm_core(A, W, D_MODEL, m0, n0, smem, acc, lane, wm, wn);

    const int nl = lane & 15, rb8 = (lane >> 4) * 8;
    #pragma unroll
    for (int s = 0; s < 2; ++s) {
        #pragma unroll
        for (int t = 0; t < 4; ++t) {
            const int n  = n0 + wn * 64 + 16 * t + nl;
            const int mb = m0 + wm * 32 + 16 * s + rb8;
            const float bv = bias[n];
            #pragma unroll
            for (int r = 0; r < 8; ++r) {
                const float v = acc[s][t][r] + bv;
                const int m = mb + r;
                if (n < D_INNER) xs[(size_t)m * D_INNER + n] = v;
                else             res[(size_t)m * D_INNER + (n - D_INNER)] = f2bf(v);
            }
        }
    }
}

// ---------------- fused depthwise conv + SiLU + channel mean ----------------
__global__ __launch_bounds__(256) void k_conv_silu(
    const float* __restrict__ xs, const float* __restrict__ conv_w,
    const float* __restrict__ conv_b, uint16_t* __restrict__ xsilu,
    float* __restrict__ xt_mean)
{
    const int p = blockIdx.x;                 // position = b*SEQ + s
    const int b = p / SEQ, s = p % SEQ;
    __shared__ float red[256];
    float lsum = 0.f;
    for (int c = threadIdx.x; c < D_INNER; c += 256) {
        float acc = conv_b[c];
        #pragma unroll
        for (int k = 0; k < D_CONV; ++k) {
            const int t = s - (D_CONV - 1) + k;
            if (t >= 0) acc += xs[((size_t)b * SEQ + t) * D_INNER + c] * conv_w[c * D_CONV + k];
        }
        const float v = acc / (1.f + __expf(-acc));   // SiLU
        xsilu[(size_t)p * D_INNER + c] = f2bf(v);
        lsum += v;
    }
    red[threadIdx.x] = lsum;
    __syncthreads();
    for (int off = 128; off > 0; off >>= 1) {
        if ((int)threadIdx.x < off) red[threadIdx.x] += red[threadIdx.x + off];
        __syncthreads();
    }
    if (threadIdx.x == 0) xt_mean[p] = red[0] * (1.f / D_INNER);
}

// ---------------- GEMM2: x_ssm = xs_silu @ W_x^T + b_x (N padded to 128) ----------------
__global__ __launch_bounds__(256) void k_gemm2(
    const uint16_t* __restrict__ A,   // NPOS x D_INNER bf16
    const uint16_t* __restrict__ W,   // N_SSM_PAD x D_INNER bf16 (pad rows zero)
    const float*    __restrict__ bias,
    float*          __restrict__ C)   // NPOS x N_SSM_PAD f32
{
    __shared__ uint16_t smem[4 * TILE_ELMS];
    const int lane = threadIdx.x & 31, wave = threadIdx.x >> 5;
    const int wm = wave & 3, wn = wave >> 2;
    const int m0 = blockIdx.y * M_BLK, n0 = blockIdx.x * N_BLK;
    v8f acc[2][4] = {};
    gemm_core(A, W, D_INNER, m0, n0, smem, acc, lane, wm, wn);

    const int nl = lane & 15, rb8 = (lane >> 4) * 8;
    #pragma unroll
    for (int s = 0; s < 2; ++s) {
        #pragma unroll
        for (int t = 0; t < 4; ++t) {
            const int n  = n0 + wn * 64 + 16 * t + nl;
            const int mb = m0 + wm * 32 + 16 * s + rb8;
            const float bv = bias[n];
            #pragma unroll
            for (int r = 0; r < 8; ++r)
                C[(size_t)(mb + r) * N_SSM_PAD + n] = acc[s][t][r] + bv;
        }
    }
}

// ---------------- delta_t = mean_i softplus(delta_raw * W_dt[i] + b_dt[i]) ----------------
__global__ __launch_bounds__(256) void k_delta(
    const float* __restrict__ xssm, const float* __restrict__ W_dt,
    const float* __restrict__ b_dt, float* __restrict__ delta_t)
{
    const int lane = threadIdx.x & 31;
    const int wave = threadIdx.x >> 5;
    const int p = blockIdx.x * 8 + wave;
    const float dr = xssm[(size_t)p * N_SSM_PAD + 2 * D_STATE];
    float lsum = 0.f;
    for (int i = lane; i < D_INNER; i += 32) {
        const float z = dr * W_dt[i] + b_dt[i];
        lsum += fmaxf(z, 0.f) + log1pf(__expf(-fabsf(z)));   // softplus
    }
    #pragma unroll
    for (int off = 16; off; off >>= 1) lsum += __shfl_xor(lsum, off, 32);
    if (lane == 0) delta_t[p] = lsum * (1.f / D_INNER);
}

// ---------------- sequential scan: one wave32 per batch, lane = state index ----------------
__global__ __launch_bounds__(256) void k_scan(
    const float* __restrict__ A_log, const float* __restrict__ xssm,
    const float* __restrict__ xt_mean, const float* __restrict__ delta_t,
    float* __restrict__ ys)
{
    const int lane = threadIdx.x & 31;   // state index i
    const int b    = threadIdx.x >> 5;   // batch
    float a_row[D_STATE];
    #pragma unroll
    for (int j = 0; j < D_STATE; ++j) a_row[j] = -__expf(A_log[lane * D_STATE + j]);
    float y = 0.f;
    for (int s = 0; s < SEQ; ++s) {
        const size_t p = (size_t)b * SEQ + s;
        const float dm = delta_t[p];
        const float xm = xt_mean[p];
        const float Bt = xssm[p * N_SSM_PAD + lane];
        const float Ct = xssm[p * N_SSM_PAD + D_STATE + lane];
        float acc = 0.f;
        #pragma unroll
        for (int j = 0; j < D_STATE; ++j)
            acc += __expf(dm * a_row[j]) * __shfl(y, j, 32);
        const float yn = acc + Bt * xm;
        float n2 = yn * yn;
        #pragma unroll
        for (int off = 16; off; off >>= 1) n2 += __shfl_xor(n2, off, 32);
        y = yn / (sqrtf(n2) + 1e-6f);
        float o = Ct * y;
        #pragma unroll
        for (int off = 16; off; off >>= 1) o += __shfl_xor(o, off, 32);
        if (lane == 0) ys[p] = o;
    }
}

// ---------------- GEMM3: out = res @ W_out^T + b_out + ys*rowsum(W_out) ----------------
__global__ __launch_bounds__(256) void k_gemm3(
    const uint16_t* __restrict__ A,    // NPOS x D_INNER bf16 (res)
    const uint16_t* __restrict__ W,    // D_OUT x D_INNER bf16
    const float*    __restrict__ b_out,
    const float*    __restrict__ rowsum,
    const float*    __restrict__ ys,
    float*          __restrict__ out)  // NPOS x D_OUT f32
{
    __shared__ uint16_t smem[4 * TILE_ELMS];
    const int lane = threadIdx.x & 31, wave = threadIdx.x >> 5;
    const int wm = wave & 3, wn = wave >> 2;
    const int m0 = blockIdx.y * M_BLK, n0 = blockIdx.x * N_BLK;
    v8f acc[2][4] = {};
    gemm_core(A, W, D_INNER, m0, n0, smem, acc, lane, wm, wn);

    const int nl = lane & 15, rb8 = (lane >> 4) * 8;
    #pragma unroll
    for (int s = 0; s < 2; ++s) {
        #pragma unroll
        for (int t = 0; t < 4; ++t) {
            const int n  = n0 + wn * 64 + 16 * t + nl;
            const int mb = m0 + wm * 32 + 16 * s + rb8;
            const float bv = b_out[n];
            const float rs = rowsum[n];
            #pragma unroll
            for (int r = 0; r < 8; ++r) {
                const int m = mb + r;
                out[(size_t)m * D_OUT + n] = acc[s][t][r] + bv + ys[m] * rs;
            }
        }
    }
}

// ---------------- host launcher ----------------
extern "C" void kernel_launch(void* const* d_in, const int* in_sizes, int n_in,
                              void* d_out, int out_size, void* d_ws, size_t ws_size,
                              hipStream_t stream)
{
    const float* x      = (const float*)d_in[0];
    const float* W_in   = (const float*)d_in[1];
    const float* b_in   = (const float*)d_in[2];
    const float* conv_w = (const float*)d_in[3];
    const float* conv_b = (const float*)d_in[4];
    const float* W_x    = (const float*)d_in[5];
    const float* b_x    = (const float*)d_in[6];
    const float* W_dt   = (const float*)d_in[7];
    const float* b_dt   = (const float*)d_in[8];
    const float* W_out  = (const float*)d_in[9];
    const float* b_out  = (const float*)d_in[10];
    const float* A_log  = (const float*)d_in[11];
    float* out = (float*)d_out;

    char*  ws  = (char*)d_ws;
    size_t off = 0;
    auto alloc = [&](size_t bytes) -> void* {
        void* p = ws + off;
        off = (off + bytes + 255) & ~(size_t)255;
        return p;
    };
    uint16_t* x_bf    = (uint16_t*)alloc((size_t)NPOS * D_MODEL * 2);
    uint16_t* win_bf  = (uint16_t*)alloc((size_t)2 * D_INNER * D_MODEL * 2);
    uint16_t* wx_bf   = (uint16_t*)alloc((size_t)N_SSM_PAD * D_INNER * 2);
    float*    bx_p    = (float*)   alloc(N_SSM_PAD * 4);
    uint16_t* wout_bf = (uint16_t*)alloc((size_t)D_OUT * D_INNER * 2);
    float*    rowsum  = (float*)   alloc(D_OUT * 4);
    float*    xs      = (float*)   alloc((size_t)NPOS * D_INNER * 4);
    uint16_t* res_bf  = (uint16_t*)alloc((size_t)NPOS * D_INNER * 2);
    uint16_t* xsilu   = (uint16_t*)alloc((size_t)NPOS * D_INNER * 2);
    float*    xssm    = (float*)   alloc((size_t)NPOS * N_SSM_PAD * 4);
    float*    xtmean  = (float*)   alloc((size_t)NPOS * 4);
    float*    dtt     = (float*)   alloc((size_t)NPOS * 4);
    float*    ysbuf   = (float*)   alloc((size_t)NPOS * 4);

    // precision conversion + weight prep
    {
        int n = NPOS * D_MODEL;
        k_f2bf<<<(n + 255) / 256, 256, 0, stream>>>(x, x_bf, n);
    }
    {
        int n = 2 * D_INNER * D_MODEL;
        k_f2bf<<<(n + 255) / 256, 256, 0, stream>>>(W_in, win_bf, n);
    }
    {
        int n = D_OUT * D_INNER;
        k_f2bf<<<(n + 255) / 256, 256, 0, stream>>>(W_out, wout_bf, n);
    }
    k_pad_wx<<<(N_SSM_PAD * D_INNER + 255) / 256, 256, 0, stream>>>(W_x, b_x, wx_bf, bx_p);
    k_rowsum<<<1, 128, 0, stream>>>(W_out, rowsum);

    // pipeline
    k_gemm1<<<dim3(2 * D_INNER / N_BLK, NPOS / M_BLK), 256, 0, stream>>>(x_bf, win_bf, b_in, xs, res_bf);
    k_conv_silu<<<NPOS, 256, 0, stream>>>(xs, conv_w, conv_b, xsilu, xtmean);
    k_gemm2<<<dim3(N_SSM_PAD / N_BLK, NPOS / M_BLK), 256, 0, stream>>>(xsilu, wx_bf, bx_p, xssm);
    k_delta<<<NPOS / 8, 256, 0, stream>>>(xssm, W_dt, b_dt, dtt);
    k_scan<<<1, 256, 0, stream>>>(A_log, xssm, xtmean, dtt, ysbuf);
    k_gemm3<<<dim3(D_OUT / N_BLK, NPOS / M_BLK), 256, 0, stream>>>(res_bf, wout_bf, b_out, rowsum, ysbuf, out);
}
